// RGBReconstructionModel_30262339567878
// MI455X (gfx1250) — compile-verified, matrix operand order checked by reading, not run on hardware
//
#include <hip/hip_runtime.h>

typedef __attribute__((ext_vector_type(16))) _Float16 v16h;
typedef __attribute__((ext_vector_type(8)))  float    v8f;

#define B_PTS   262144
#define NLVL    24
#define TSZ     (1u << 22)
#define HMASK   (TSZ - 1u)
#define P2H     2654435761u
#define P3H     805459861u

// ---------------------------------------------------------------------------
// Kernel 0: convert f32 weights to f16, packed column-major [n][k] so a lane's
// B-matrix fragment (16 halves for one 32-K chunk) is two contiguous 16B loads.
// ---------------------------------------------------------------------------
__global__ void pack_w_kernel(const float* __restrict__ W, _Float16* __restrict__ Wp,
                              int K, int N, int Npad) {
    int i = blockIdx.x * blockDim.x + threadIdx.x;
    int total = K * Npad;
    if (i >= total) return;
    int n = i / K;
    int k = i - n * K;
    float v = (n < N) ? W[k * N + n] : 0.0f;
    Wp[n * K + k] = (_Float16)v;
}

// ---------------------------------------------------------------------------
// Kernel 1: hash-grid encode. One thread per point; 24 spatial levels
// (8 corners, trilinear) + 24 temporal levels (2 taps, linear).
// Emits B x 96 f16 feature matrix (row-major), ready for WMMA A-fragments.
// ---------------------------------------------------------------------------
__global__ void __launch_bounds__(256)
encode_kernel(const float* __restrict__ coords,
              const float* __restrict__ stab,
              const float* __restrict__ ttab,
              _Float16* __restrict__ feat) {
    int i = blockIdx.x * blockDim.x + threadIdx.x;
    float4 c4 = ((const float4*)coords)[i];
    const float x = c4.x, y = c4.y, z = c4.z, t = c4.w;
    _Float16* fo = feat + (size_t)i * 96;

    float res = 16.0f;   // RES[l] = floor(16 * 1.5^l)
    #pragma unroll 1
    for (int l = 0; l < NLVL; ++l) {
        const float r = floorf(res);
        // ---- spatial ----
        float sx = x * r, sy = y * r, sz = z * r;
        float bx = floorf(sx), by = floorf(sy), bz = floorf(sz);
        float fx = sx - bx, fy = sy - by, fz = sz - bz;
        unsigned ix = (unsigned)(int)bx;
        unsigned iy = (unsigned)(int)by;
        unsigned iz = (unsigned)(int)bz;
        unsigned hx0 = ix,        hx1 = ix + 1u;       // P1 == 1
        unsigned hy0 = iy * P2H,  hy1 = hy0 + P2H;
        unsigned hz0 = iz * P3H,  hz1 = hz0 + P3H;
        const float2* tab = (const float2*)stab + (size_t)l * TSZ;
        float f0 = 0.0f, f1 = 0.0f;
        #pragma unroll
        for (int c = 0; c < 8; ++c) {
            const int ox = (c >> 2) & 1, oy = (c >> 1) & 1, oz = c & 1;
            unsigned h = ((ox ? hx1 : hx0) ^ (oy ? hy1 : hy0) ^ (oz ? hz1 : hz0)) & HMASK;
            float2 v = tab[h];
            float w = (ox ? fx : 1.0f - fx) * (oy ? fy : 1.0f - fy) * (oz ? fz : 1.0f - fz);
            f0 = fmaf(w, v.x, f0);
            f1 = fmaf(w, v.y, f1);
        }
        fo[2 * l]     = (_Float16)f0;
        fo[2 * l + 1] = (_Float16)f1;
        // ---- temporal ----
        float st = t * r;
        float bt = floorf(st);
        float ft = st - bt;
        int i0 = (int)bt;
        unsigned h0 = ((unsigned)i0 * P2H) & HMASK;
        unsigned h1 = ((unsigned)(i0 + 1) * P2H) & HMASK;
        const float2* tt = (const float2*)ttab + (size_t)l * TSZ;
        float2 v0 = tt[h0];
        float2 v1 = tt[h1];
        fo[48 + 2 * l]     = (_Float16)((1.0f - ft) * v0.x + ft * v1.x);
        fo[48 + 2 * l + 1] = (_Float16)((1.0f - ft) * v0.y + ft * v1.y);
        res *= 1.5f;
    }
}

// ---------------------------------------------------------------------------
// Kernel 2: fused MLP via v_wmma_f32_16x16x32_f16.
// 256 threads = 8 waves; each wave computes a 16-row tile end-to-end.
// Activations staged in LDS with 16B-block XOR swizzle (conflict-free
// ds_load_b128 for A-fragments). LDS = 8*16*256*2 = 64KB.
// ---------------------------------------------------------------------------

// CDNA5 16-bit A/B operand per-lane layout for one 32-K chunk:
// lanes 0-15: K {base+0..7, base+16..23}; lanes 16-31: K {base+8..15, base+24..31}.
// `base` already includes the hi*8 shift; each run is one contiguous 16B load.
__device__ __forceinline__ v16h load_frag_g(const _Float16* p) {
    v16h a;
    ((float4*)&a)[0] = *(const float4*)(p);
    ((float4*)&a)[1] = *(const float4*)(p + 16);
    return a;
}

__device__ __forceinline__ v16h load_frag_lds(const _Float16* rowbase, int b0, int r) {
    v16h a;
    ((float4*)&a)[0] = *(const float4*)(rowbase + (((b0    ) ^ r) << 3));
    ((float4*)&a)[1] = *(const float4*)(rowbase + (((b0 + 2) ^ r) << 3));
    return a;
}

__device__ __forceinline__ void store_swz(_Float16* rowbase, int r, int col, float v) {
    int blk = ((col >> 3) ^ r);
    rowbase[(blk << 3) | (col & 7)] = (_Float16)v;
}

__global__ void __launch_bounds__(256)
mlp_kernel(const _Float16* __restrict__ feat,
           const _Float16* __restrict__ W1p, const float* __restrict__ b1,
           const _Float16* __restrict__ W2p, const float* __restrict__ b2,
           const _Float16* __restrict__ W3p, const float* __restrict__ b3,
           float* __restrict__ out) {
    __shared__ _Float16 hbuf[8][16][256];
    const int wave = threadIdx.x >> 5;
    const int lane = threadIdx.x & 31;
    const int lr   = lane & 15;   // A row / B,C column within tile
    const int hi   = lane >> 4;   // K-phase select (0 or 1)
    const int row0 = blockIdx.x * 128 + wave * 16;
    _Float16 (*h1)[256] = hbuf[wave];

    // ---------------- layer 1: [16x96] x [96x256] ----------------
    v16h a1[3];
    {
        const _Float16* fr = feat + (size_t)(row0 + lr) * 96 + hi * 8;
        #pragma unroll
        for (int c = 0; c < 3; ++c) a1[c] = load_frag_g(fr + c * 32);
    }
    #pragma unroll 1
    for (int nt = 0; nt < 16; ++nt) {
        v8f acc = {};
        const _Float16* wcol = W1p + (nt * 16 + lr) * 96 + hi * 8;
        #pragma unroll
        for (int c = 0; c < 3; ++c) {
            v16h b = load_frag_g(wcol + c * 32);
            acc = __builtin_amdgcn_wmma_f32_16x16x32_f16(false, a1[c], false, b,
                                                         (short)0, acc, false, false);
        }
        float bias = b1[nt * 16 + lr];
        #pragma unroll
        for (int i = 0; i < 8; ++i) {
            float v = fmaxf(acc[i] + bias, 0.0f);
            store_swz(&h1[i + hi * 8][0], i + hi * 8, nt * 16 + lr, v);
        }
    }
    __syncthreads();

    // ---------------- layer 2: [16x256] x [256x256] ----------------
    v16h a2[8];
    #pragma unroll
    for (int c = 0; c < 8; ++c)
        a2[c] = load_frag_lds(&h1[lr][0], c * 4 + hi, lr);
    #pragma unroll 1
    for (int nt = 0; nt < 16; ++nt) {
        v8f acc = {};
        const _Float16* wcol = W2p + (size_t)(nt * 16 + lr) * 256 + hi * 8;
        #pragma unroll
        for (int c = 0; c < 8; ++c) {
            v16h b = load_frag_g(wcol + c * 32);
            acc = __builtin_amdgcn_wmma_f32_16x16x32_f16(false, a2[c], false, b,
                                                         (short)0, acc, false, false);
        }
        float bias = b2[nt * 16 + lr];
        #pragma unroll
        for (int i = 0; i < 8; ++i) {
            float v = fmaxf(acc[i] + bias, 0.0f);
            store_swz(&h1[i + hi * 8][0], i + hi * 8, nt * 16 + lr, v);
        }
    }
    __syncthreads();

    // ---------------- layer 3: [16x256] x [256x16] (3 live cols) ----------------
    v16h a3[8];
    #pragma unroll
    for (int c = 0; c < 8; ++c)
        a3[c] = load_frag_lds(&h1[lr][0], c * 4 + hi, lr);
    v8f acc = {};
    {
        const _Float16* wcol = W3p + lr * 256 + hi * 8;
        #pragma unroll
        for (int c = 0; c < 8; ++c) {
            v16h b = load_frag_g(wcol + c * 32);
            acc = __builtin_amdgcn_wmma_f32_16x16x32_f16(false, a3[c], false, b,
                                                         (short)0, acc, false, false);
        }
    }
    if (lr < 3) {
        float bias = b3[lr];
        #pragma unroll
        for (int i = 0; i < 8; ++i) {
            int row = row0 + i + hi * 8;
            float v = acc[i] + bias;
            out[row * 3 + lr] = 1.0f / (1.0f + __expf(-v));
        }
    }
}

// ---------------------------------------------------------------------------
extern "C" void kernel_launch(void* const* d_in, const int* in_sizes, int n_in,
                              void* d_out, int out_size, void* d_ws, size_t ws_size,
                              hipStream_t stream) {
    (void)in_sizes; (void)n_in; (void)out_size; (void)ws_size;
    const float* coords = (const float*)d_in[0];
    const float* stab   = (const float*)d_in[1];
    const float* ttab   = (const float*)d_in[2];
    const float* W1     = (const float*)d_in[3];
    const float* b1     = (const float*)d_in[4];
    const float* W2     = (const float*)d_in[5];
    const float* b2     = (const float*)d_in[6];
    const float* W3     = (const float*)d_in[7];
    const float* b3     = (const float*)d_in[8];

    char* ws = (char*)d_ws;
    _Float16* feat = (_Float16*)ws;                                   // B*96 f16 = 48 MB
    _Float16* W1p  = (_Float16*)(ws + 50331648);                      // 256*96
    _Float16* W2p  = (_Float16*)(ws + 50331648 + 49152);              // 256*256
    _Float16* W3p  = (_Float16*)(ws + 50331648 + 49152 + 131072);     // 16*256

    pack_w_kernel<<<(96 * 256 + 255) / 256, 256, 0, stream>>>(W1, W1p, 96, 256, 256);
    pack_w_kernel<<<(256 * 256 + 255) / 256, 256, 0, stream>>>(W2, W2p, 256, 256, 256);
    pack_w_kernel<<<(256 * 16 + 255) / 256, 256, 0, stream>>>(W3, W3p, 256, 3, 16);
    encode_kernel<<<B_PTS / 256, 256, 0, stream>>>(coords, stab, ttab, feat);
    mlp_kernel<<<B_PTS / 128, 256, 0, stream>>>(feat, W1p, b1, W2p, b2, W3p, b3,
                                                (float*)d_out);
}
